// HGNNPConv_40303973106024
// MI455X (gfx1250) — matrix operand order repaired and checked by reading, not compiled, not run on hardware
//
#include <hip/hip_runtime.h>
#include <hip/hip_bf16.h>

// HGNN+ conv: Xt = X@W^T + b ; Xe = (H^T@Xt)/De ; out = relu((H@Xe)/Dv)
// Heavy GEMMs via v_wmma_f32_16x16x32_bf16 (f32 accumulate), f32->bf16
// conversion during LDS staging. Double-buffered LDS: next chunk's global
// loads are issued before the WMMAs so HBM latency hides under compute;
// global_prefetch_b8 keeps chunk k+2 in flight. HBM-bound target:
// H read 3x (degree pass + 2 GEMMs) ~= 1.5 GB @ 23.3 TB/s.

typedef __bf16 bf16;
typedef __attribute__((ext_vector_type(16))) __bf16 v16bf;
typedef __attribute__((ext_vector_type(8)))  __bf16 v8bf;
typedef __attribute__((ext_vector_type(8)))  float  v8f;

#define N_V 16384
#define N_E 8192
#define CH  512

#define BM 128
#define BN 128
#define BK 32
#define SA 40   // padded LDS row stride (elements) for As[BM][BK]
#define SB 40   // padded LDS row stride (elements) for Bs[BN][BK]

// ---------------------------------------------------------------------------
// Degree pass: one read of H computes Dv[v] = sum_e H[v,e] and De[e] = sum_v.
// 128 rows per block; each thread owns 32 columns (t + 256*j).
// ---------------------------------------------------------------------------
__global__ __launch_bounds__(256) void degrees_kernel(const float* __restrict__ H,
                                                      float* __restrict__ Dv,
                                                      float* __restrict__ De) {
    __shared__ float rowsum[128];
    const int t  = threadIdx.x;
    const int r0 = blockIdx.x * 128;

    for (int i = t; i < 128; i += 256) rowsum[i] = 0.f;
    __syncthreads();

    float colacc[32];
#pragma unroll
    for (int j = 0; j < 32; ++j) colacc[j] = 0.f;

    for (int rr = 0; rr < 128; ++rr) {
        const float* row = H + (size_t)(r0 + rr) * N_E;
        float rp = 0.f;
#pragma unroll
        for (int j = 0; j < 32; ++j) {
            float v = row[t + 256 * j];
            colacc[j] += v;
            rp += v;
        }
#pragma unroll
        for (int off = 16; off > 0; off >>= 1) rp += __shfl_xor(rp, off, 32);
        if ((t & 31) == 0) atomicAdd(&rowsum[rr], rp);
    }
    __syncthreads();

    for (int i = t; i < 128; i += 256) Dv[r0 + i] = rowsum[i];  // rows unique per block
#pragma unroll
    for (int j = 0; j < 32; ++j) atomicAdd(&De[t + 256 * j], colacc[j]);
}

// ---------------------------------------------------------------------------
// Generic bf16-WMMA block GEMM: C[M,N] = op(A) * op(B) (+bias) (*1/deg) (relu)
//   TA==0: A row-major [m][k] ; TA==1: A(m,k) = Ag[k*lda + m] (A = Hg^T)
//   TB==0: B already [n][k] (e.g. W) ; TB==1: B row-major [k][n], transposed
// Block: 256 threads, 128x128 tile; wave (4x2 grid) owns 32x64 = 2x4 tiles.
// Double-buffered LDS, one barrier per K-chunk.
// ---------------------------------------------------------------------------
template <int TA, int TB, int BIAS, int RSCALE, int RELU>
__global__ __launch_bounds__(256) void gemm_bf16_wmma(
    const float* __restrict__ A, int lda,
    const float* __restrict__ B, int ldb,
    const float* __restrict__ bias,    // [N]   (BIAS)
    const float* __restrict__ rowdeg,  // [M]   (RSCALE, safe reciprocal)
    float* __restrict__ C, int ldc,
    int K)
{
    __shared__ __align__(16) bf16 As[2][BM * SA];
    __shared__ __align__(16) bf16 Bs[2][BN * SB];

    const int t    = threadIdx.x;
    const int lane = t & 31;
    const int wid  = t >> 5;
    const int wm   = wid & 3;   // 4 waves along M, 32 rows each
    const int wn   = wid >> 2;  // 2 waves along N, 64 cols each
    const int m0   = blockIdx.y * BM;
    const int n0   = blockIdx.x * BN;

    const int lr = lane & 15;   // row (A) / col (B) within 16
    const int lh = lane >> 4;   // half-wave selector
    const int nk = K / BK;

    v8f acc[2][4];
#pragma unroll
    for (int i = 0; i < 2; ++i)
#pragma unroll
        for (int j = 0; j < 4; ++j) acc[i][j] = (v8f){0.f,0.f,0.f,0.f,0.f,0.f,0.f,0.f};

    // ---- prologue: stage chunk 0 into buffer 0 ----
    if (TA == 0) {
#pragma unroll
        for (int j = 0; j < 16; ++j) {
            int i = t + 256 * j; int r = i >> 5, c = i & 31;
            As[0][r * SA + c] = (bf16)A[(size_t)(m0 + r) * lda + c];
        }
    } else {
#pragma unroll
        for (int j = 0; j < 16; ++j) {
            int i = t + 256 * j; int kr = i >> 7, mc = i & 127;
            As[0][mc * SA + kr] = (bf16)A[(size_t)kr * lda + (m0 + mc)];
        }
    }
    if (TB == 0) {
#pragma unroll
        for (int j = 0; j < 16; ++j) {
            int i = t + 256 * j; int r = i >> 5, c = i & 31;
            Bs[0][r * SB + c] = (bf16)B[(size_t)(n0 + r) * ldb + c];
        }
    } else {
#pragma unroll
        for (int j = 0; j < 16; ++j) {
            int i = t + 256 * j; int kr = i >> 7, nc = i & 127;
            Bs[0][nc * SB + kr] = (bf16)B[(size_t)kr * ldb + (n0 + nc)];
        }
    }
    __syncthreads();

    float ra[16], rb[16];

    for (int kc = 0; kc < nk; ++kc) {
        const int cur = kc & 1;
        const bf16* Ac = As[cur];
        const bf16* Bc = Bs[cur];
        const bool havenext = (kc + 1) < nk;

        // ---- issue next chunk's global loads (in flight under the WMMAs) ----
        if (havenext) {
            const int k0 = (kc + 1) * BK;
            if (TA == 0) {
#pragma unroll
                for (int j = 0; j < 16; ++j) {
                    int i = t + 256 * j; int r = i >> 5, c = i & 31;
                    ra[j] = A[(size_t)(m0 + r) * lda + (k0 + c)];
                }
            } else {
#pragma unroll
                for (int j = 0; j < 16; ++j) {
                    int i = t + 256 * j; int kr = i >> 7, mc = i & 127;
                    ra[j] = A[(size_t)(k0 + kr) * lda + (m0 + mc)];
                }
            }
            if (TB == 0) {
#pragma unroll
                for (int j = 0; j < 16; ++j) {
                    int i = t + 256 * j; int r = i >> 5, c = i & 31;
                    rb[j] = B[(size_t)(n0 + r) * ldb + (k0 + c)];
                }
            } else {
#pragma unroll
                for (int j = 0; j < 16; ++j) {
                    int i = t + 256 * j; int kr = i >> 7, nc = i & 127;
                    rb[j] = B[(size_t)(k0 + kr) * ldb + (n0 + nc)];
                }
            }
        }
        // ---- prefetch chunk kc+2 (global_prefetch_b8) ----
        if (kc + 2 < nk) {
            const int k2 = (kc + 2) * BK;
            if (TA == 0) __builtin_prefetch(&A[(size_t)(m0 + (t >> 1)) * lda + k2 + (t & 1) * 16], 0, 3);
            else         __builtin_prefetch(&A[(size_t)(k2 + (t >> 3)) * lda + m0 + (t & 7) * 16], 0, 3);
            if (TB == 0) __builtin_prefetch(&B[(size_t)(n0 + (t >> 1)) * ldb + k2 + (t & 1) * 16], 0, 3);
            else         __builtin_prefetch(&B[(size_t)(k2 + (t >> 3)) * ldb + n0 + (t & 7) * 16], 0, 3);
        }

        // ---- fragments per ISA 16-bit A(16x32)/B(32x16) layouts + WMMA ----
        v16bf af[2], bfr[4];
#pragma unroll
        for (int i = 0; i < 2; ++i) {
            // lane<16: K{0..7,16..23}; lane>=16: K{8..15,24..31}
            const bf16* p = &Ac[(wm * 32 + i * 16 + lr) * SA + lh * 8];
            union { v16bf v; v8bf x[2]; } u;
            u.x[0] = *(const v8bf*)(p);
            u.x[1] = *(const v8bf*)(p + 16);
            af[i] = u.v;
        }
#pragma unroll
        for (int j = 0; j < 4; ++j) {
            // lane<16: K{0..7,16..23} of column lr; lane>=16: K{8..15,24..31}
            const bf16* p = &Bc[(wn * 64 + j * 16 + lr) * SB + lh * 8];
            union { v16bf v; v8bf x[2]; } u;
            u.x[0] = *(const v8bf*)(p);
            u.x[1] = *(const v8bf*)(p + 16);
            bfr[j] = u.v;
        }

#pragma unroll
        for (int i = 0; i < 2; ++i)
#pragma unroll
            for (int j = 0; j < 4; ++j)
                acc[i][j] = __builtin_amdgcn_wmma_f32_16x16x32_bf16(
                    false, af[i], false, bfr[j], (short)0, acc[i][j], false, false);

        // ---- convert + store the staged registers into the other buffer ----
        if (havenext) {
            bf16* An = As[cur ^ 1];
            bf16* Bn = Bs[cur ^ 1];
            if (TA == 0) {
#pragma unroll
                for (int j = 0; j < 16; ++j) {
                    int i = t + 256 * j; int r = i >> 5, c = i & 31;
                    An[r * SA + c] = (bf16)ra[j];
                }
            } else {
#pragma unroll
                for (int j = 0; j < 16; ++j) {
                    int i = t + 256 * j; int kr = i >> 7, mc = i & 127;
                    An[mc * SA + kr] = (bf16)ra[j];
                }
            }
            if (TB == 0) {
#pragma unroll
                for (int j = 0; j < 16; ++j) {
                    int i = t + 256 * j; int r = i >> 5, c = i & 31;
                    Bn[r * SB + c] = (bf16)rb[j];
                }
            } else {
#pragma unroll
                for (int j = 0; j < 16; ++j) {
                    int i = t + 256 * j; int kr = i >> 7, nc = i & 127;
                    Bn[nc * SB + kr] = (bf16)rb[j];
                }
            }
        }
        __syncthreads();
    }

    // ---- epilogue: D layout = VGPR e, lane<16 -> (M=e,N=lane); lane>=16 -> (M=e+8) ----
#pragma unroll
    for (int i = 0; i < 2; ++i) {
#pragma unroll
        for (int j = 0; j < 4; ++j) {
            const int n  = n0 + wn * 64 + j * 16 + lr;
            const int mb = m0 + wm * 32 + i * 16 + lh * 8;
            const float bn = BIAS ? bias[n] : 0.f;
#pragma unroll
            for (int e = 0; e < 8; ++e) {
                const int m = mb + e;
                float v = acc[i][j][e] + bn;
                if (RSCALE) { float d = rowdeg[m]; v = (d == 0.f) ? 0.f : v / d; }
                if (RELU) v = fmaxf(v, 0.f);
                C[(size_t)m * ldc + n] = v;
            }
        }
    }
}

// ---------------------------------------------------------------------------
extern "C" void kernel_launch(void* const* d_in, const int* in_sizes, int n_in,
                              void* d_out, int out_size, void* d_ws, size_t ws_size,
                              hipStream_t stream) {
    const float* X = (const float*)d_in[0];   // [N_V, CH]
    const float* H = (const float*)d_in[1];   // [N_V, N_E]
    const float* W = (const float*)d_in[2];   // [CH, CH] (out, in)
    const float* b = (const float*)d_in[3];   // [CH]
    float* out = (float*)d_out;               // [N_V, CH]

    float* Xt = (float*)d_ws;                  // N_V*CH
    float* Xe = Xt + (size_t)N_V * CH;         // N_E*CH
    float* De = Xe + (size_t)N_E * CH;         // N_E
    float* Dv = De + N_E;                      // N_V

    hipMemsetAsync(De, 0, (size_t)(N_E + N_V) * sizeof(float), stream);
    degrees_kernel<<<N_V / 128, 256, 0, stream>>>(H, Dv, De);

    // Xt = X @ W^T + b : A=X [m][k], B=W already [n][k]
    gemm_bf16_wmma<0, 0, 1, 0, 0><<<dim3(CH / BN, N_V / BM), 256, 0, stream>>>(
        X, CH, W, CH, b, nullptr, Xt, CH, CH);

    // Xe = (H^T @ Xt) / De : A=(m=e,k=v) = H[v][e] (TA=1, lda=N_E), B=Xt [k][n] (TB=1)
    gemm_bf16_wmma<1, 1, 0, 1, 0><<<dim3(CH / BN, N_E / BM), 256, 0, stream>>>(
        H, N_E, Xt, CH, nullptr, De, Xe, CH, N_V);

    // out = relu((H @ Xe) / Dv) : A=H [m=v][k=e], B=Xe [k][n] (TB=1)
    gemm_bf16_wmma<0, 1, 0, 1, 1><<<dim3(CH / BN, N_V / BM), 256, 0, stream>>>(
        H, N_E, Xe, CH, nullptr, Dv, out, CH, N_E);
}
